// DeformableCrossAttention2D_Q_84189948936664
// MI455X (gfx1250) — matrix operand order repaired and checked by reading, not compiled, not run on hardware
//
#include <hip/hip_runtime.h>
#include <hip/hip_bf16.h>
#include <math.h>

// ---------------------------------------------------------------------------
// Problem constants (from the reference)
// ---------------------------------------------------------------------------
#define KQ   6
#define BQ   8
#define TQ   12
#define DIM  256
#define HALF 256
#define NH   8
#define G    8
#define BS   (KQ * BQ * TQ)      // 576 queries
#define NS   (BS * G)            // 4608 samples
#define HBEV 256
#define WBEV 256
#define NORM_F 0.01953125f       // 2/102.4
#define ATT_SCALE 0.125f         // 64^-0.5

typedef __attribute__((ext_vector_type(16))) __bf16 v16bf;
typedef __attribute__((ext_vector_type(8)))  float  v8f;

// ---------------------------------------------------------------------------
// WMMA helpers: D = A(16x256 f32, LDS) @ W(256x256, pre-swizzled bf16) + bias
// Block = 256 threads = 8 waves; wave w owns N-tiles 2w and 2w+1.
// A-fragment layout (16-bit A 16x32; contiguous k order used here, not the ISA table order):
//   lanes 0-15 : row=lane,    elems 0-15 -> K 0-15
//   lanes 16-31: row=lane-16, elems 0-15 -> K 16-31
// B pre-swizzle (dense 32x16 bf16 B): lane holds col N=lane&15,
//   elem j -> K = (lane<16 ? 0 : 16) + j.
// D layout: vgpr r, lane l -> M = r + (l&16 ? 8 : 0), N = l&15.
// ---------------------------------------------------------------------------
__device__ __forceinline__ v16bf make_afrag(const float* __restrict__ A, int lane) {
  const int   row = lane & 15;
  const int   kb  = (lane & 16) ? 16 : 0;
  const float* p  = A + row * 256;
  v16bf a;
#pragma unroll
  for (int j = 0; j < 8; ++j) a[j]     = (__bf16)p[kb + j];
#pragma unroll
  for (int j = 0; j < 8; ++j) a[j + 8] = (__bf16)p[kb + 8 + j];
  return a;
}

// act: 0 = none, 1 = relu
__device__ __forceinline__ void gemm_16x256(const float* __restrict__ Alds,
                                            const __bf16* __restrict__ packW,
                                            const float* __restrict__ bias,
                                            float* __restrict__ Clds,
                                            int act) {
  const int lane = threadIdx.x & 31;
  const int wave = threadIdx.x >> 5;
  const int t0 = wave * 2, t1 = t0 + 1;
  v8f acc0 = {};
  v8f acc1 = {};
  const v16bf* bp = (const v16bf*)packW;
#pragma unroll
  for (int s = 0; s < 8; ++s) {                       // K steps of 32
    v16bf a  = make_afrag(Alds + s * 32, lane);
    v16bf b0 = bp[(t0 * 8 + s) * 32 + lane];          // 32B coalesced per lane
    v16bf b1 = bp[(t1 * 8 + s) * 32 + lane];
    acc0 = __builtin_amdgcn_wmma_f32_16x16x32_bf16(false, a, false, b0,
                                                   (short)0, acc0, false, false);
    acc1 = __builtin_amdgcn_wmma_f32_16x16x32_bf16(false, a, false, b1,
                                                   (short)0, acc1, false, false);
  }
  const int n  = lane & 15;
  const int mh = (lane & 16) ? 8 : 0;
#pragma unroll
  for (int r = 0; r < 8; ++r) {
    const int m = r + mh;
    float c0 = acc0[r];
    float c1 = acc1[r];
    if (bias) { c0 += bias[t0 * 16 + n]; c1 += bias[t1 * 16 + n]; }
    if (act == 1) { c0 = fmaxf(c0, 0.f); c1 = fmaxf(c1, 0.f); }
    Clds[m * 256 + t0 * 16 + n] = c0;
    Clds[m * 256 + t1 * 16 + n] = c1;
  }
}

// DAB-DETR sine embedding element: hidden=256, half=128, temp=40
__device__ __forceinline__ float sine_dim_scale(int dd) {
  // 2*pi / 40^(2*floor(dd/2)/128)
  const float dimt = __powf(40.0f, (float)(2 * (dd >> 1)) * (1.0f / 128.0f));
  return 6.28318530717958647692f / dimt;
}

__device__ __forceinline__ float bev_tap(const float* __restrict__ p, int x, int y) {
  const bool valid = (x >= 0) & (x < WBEV) & (y >= 0) & (y < HBEV);
  const int xi = x < 0 ? 0 : (x > WBEV - 1 ? WBEV - 1 : x);
  const int yi = y < 0 ? 0 : (y > HBEV - 1 ? HBEV - 1 : y);
  return valid ? p[yi * WBEV + xi] : 0.0f;
}

// ---------------------------------------------------------------------------
// Kernel 0: pre-swizzle eight 256x256 f32 weight matrices into bf16 B-fragments
// pack[mat][((t*8+s)*32+lane)*16+j] = W[k][n], k = s*32+(lane&16?16:0)+j,
// n = t*16+(lane&15).  8 * 65536 elements, 1 thread each.
// ---------------------------------------------------------------------------
__global__ __launch_bounds__(256) void pack_weights_bf16(
    const float* __restrict__ Wq,  const float* __restrict__ pq1,
    const float* __restrict__ pq2, const float* __restrict__ pk1,
    const float* __restrict__ pk2, const float* __restrict__ Wk,
    const float* __restrict__ Wv,  const float* __restrict__ Wo,
    __bf16* __restrict__ pack) {
  const int idx  = blockIdx.x * 256 + threadIdx.x;
  const int mat  = idx >> 16;
  const int rem  = idx & 65535;
  const int j    = rem & 15;
  const int lane = (rem >> 4) & 31;
  const int s    = (rem >> 9) & 7;
  const int t    = rem >> 12;
  const int k    = s * 32 + ((lane & 16) ? 16 : 0) + j;
  const int n    = t * 16 + (lane & 15);
  const float* W;
  switch (mat) {
    case 0: W = Wq;  break;  case 1: W = pq1; break;
    case 2: W = pq2; break;  case 3: W = pk1; break;
    case 4: W = pk2; break;  case 5: W = Wk;  break;
    case 6: W = Wv;  break;  default: W = Wo; break;
  }
  pack[idx] = (__bf16)W[k * 256 + n];
}

// ---------------------------------------------------------------------------
// Kernel 1: per-query stage.  16 queries / block (36 blocks).
//  con_q = dec@Wq+bq (WMMA) -> offset MLP (gelu, tanh) -> vgrid_scaled
//  qse sine embed -> pos_q = (relu(qse@pqW1+b1)@pqW2+b2)*query_scale (WMMA x2)
// ---------------------------------------------------------------------------
__global__ __launch_bounds__(256) void query_stage(
    const float* __restrict__ dec, const float* __restrict__ refp,
    const float* __restrict__ qscale,
    const __bf16* __restrict__ packWq, const float* __restrict__ bq,
    const float* __restrict__ offW1, const float* __restrict__ offb1,
    const float* __restrict__ offW2, const float* __restrict__ offb2,
    const __bf16* __restrict__ packPq1, const float* __restrict__ pqb1,
    const __bf16* __restrict__ packPq2, const float* __restrict__ pqb2,
    float* __restrict__ conq_ws, float* __restrict__ posq_ws,
    float* __restrict__ vgrid_ws) {
  __shared__ float bufA[16 * 256];
  __shared__ float bufB[16 * 256];
  __shared__ float rp2[16 * 2];
  const int tid = threadIdx.x;
  const int r0  = blockIdx.x * 16;

  // stage dec rows into LDS (coalesced)
#pragma unroll
  for (int i = 0; i < 16; ++i) bufA[tid + i * 256] = dec[r0 * 256 + tid + i * 256];
  if (tid < 32) rp2[tid] = refp[r0 * 2 + tid];
  __syncthreads();

  gemm_16x256(bufA, packWq, bq, bufB, 0);            // con_q -> bufB
  __syncthreads();

#pragma unroll
  for (int i = 0; i < 16; ++i) conq_ws[r0 * 256 + tid + i * 256] = bufB[tid + i * 256];

  // offset MLP: one thread per (row, g); 2nd layer fused (no h[] spill)
  if (tid < 128) {
    const int rl = tid >> 3, g = tid & 7;
    const float* oin = &bufB[rl * 256 + g * 32];
    float a20 = offb2[0], a21 = offb2[1];
    for (int o = 0; o < 64; ++o) {
      float acc = offb1[o];
#pragma unroll
      for (int i2 = 0; i2 < 32; ++i2) acc += oin[i2] * offW1[i2 * 64 + o];
      const float hgelu = 0.5f * acc * (1.0f + erff(acc * 0.70710678118654752f));
      a20 += hgelu * offW2[o * 2 + 0];
      a21 += hgelu * offW2[o * 2 + 1];
    }
    const int s = (r0 + rl) * G + g;
    vgrid_ws[s * 2 + 0] = (rp2[rl * 2 + 0] + tanhf(a20) * 4.0f) * NORM_F;
    vgrid_ws[s * 2 + 1] = (rp2[rl * 2 + 1] + tanhf(a21) * 4.0f) * NORM_F;
  }

  // qse sine embed -> bufA (dec rows no longer needed); d = tid, 16 rows
  {
    const int d  = tid;
    const int dd = d & 127;
    const float sc = sine_dim_scale(dd);
#pragma unroll 4
    for (int rl = 0; rl < 16; ++rl) {
      const float p = ((d < 128) ? rp2[rl * 2 + 1] : rp2[rl * 2 + 0]) * NORM_F;
      const float e = p * sc;
      bufA[rl * 256 + d] = (dd & 1) ? __cosf(e) : __sinf(e);
    }
  }
  __syncthreads();

  gemm_16x256(bufA, packPq1, pqb1, bufB, 1);         // hidden (relu)
  __syncthreads();
  gemm_16x256(bufB, packPq2, pqb2, bufA, 0);         // pos_q pre-scale
  __syncthreads();
#pragma unroll
  for (int i = 0; i < 16; ++i) {
    const int idx = tid + i * 256;
    posq_ws[r0 * 256 + idx] = bufA[idx] * qscale[r0 * 256 + idx];
  }
}

// ---------------------------------------------------------------------------
// Kernel 2: per-sample stage.  16 samples / block (288 blocks).
// Bilinear-interpolate the 256-dim bev feature FIRST (linearity of
// grid_sample o 1x1conv), then project with WMMA:
//  con_k = feat@Wk, v = feat@Wv, pos_k = relu(kse@pkW1+b1)@pkW2+b2
// ---------------------------------------------------------------------------
__global__ __launch_bounds__(256) void sample_stage(
    const float* __restrict__ bev, const float* __restrict__ vgrid_ws,
    const __bf16* __restrict__ packWk, const __bf16* __restrict__ packWv,
    const __bf16* __restrict__ packPk1, const float* __restrict__ pkb1,
    const __bf16* __restrict__ packPk2, const float* __restrict__ pkb2,
    float* __restrict__ conk_ws, float* __restrict__ v_ws,
    float* __restrict__ posk_ws) {
  __shared__ float bufA[16 * 256];   // interpolated features
  __shared__ float bufB[16 * 256];   // kse
  __shared__ float bufC[16 * 256];   // gemm outputs
  __shared__ float sg[16 * 2];
  const int tid = threadIdx.x;
  const int s0  = blockIdx.x * 16;
  if (tid < 32) sg[tid] = vgrid_ws[s0 * 2 + tid];
  __syncthreads();

  // bilinear gather: thread = channel, loop over the 16 samples
  {
    const int c = tid;
#pragma unroll 1
    for (int rl = 0; rl < 16; ++rl) {
      const int s = s0 + rl;
      const int q = s >> 3;               // query index, (K,B,T) order
      const int b = (q / TQ) & (BQ - 1);  // batch index
      const float* bevc = bev + ((size_t)b * 256 + c) * (size_t)(HBEV * WBEV);
      const float gx = sg[rl * 2 + 0], gy = sg[rl * 2 + 1];
      const float ix = ((gx + 1.0f) * (float)WBEV - 1.0f) * 0.5f;
      const float iy = ((gy + 1.0f) * (float)HBEV - 1.0f) * 0.5f;
      const float fx = floorf(ix), fy = floorf(iy);
      const int   x0 = (int)fx, y0 = (int)fy;
      const float wx = ix - fx, wy = iy - fy;
      const float v00 = bev_tap(bevc, x0,     y0);
      const float v01 = bev_tap(bevc, x0 + 1, y0);
      const float v10 = bev_tap(bevc, x0,     y0 + 1);
      const float v11 = bev_tap(bevc, x0 + 1, y0 + 1);
      bufA[rl * 256 + c] = v00 * (1.f - wx) * (1.f - wy) + v01 * wx * (1.f - wy)
                         + v10 * (1.f - wx) * wy          + v11 * wx * wy;
    }
  }
  // kse sine embed -> bufB
  {
    const int d  = tid;
    const int dd = d & 127;
    const float sc = sine_dim_scale(dd);
#pragma unroll 4
    for (int rl = 0; rl < 16; ++rl) {
      const float p = (d < 128) ? sg[rl * 2 + 1] : sg[rl * 2 + 0];
      const float e = p * sc;
      bufB[rl * 256 + d] = (dd & 1) ? __cosf(e) : __sinf(e);
    }
  }
  __syncthreads();

  gemm_16x256(bufA, packWk, nullptr, bufC, 0);       // con_k
  __syncthreads();
#pragma unroll
  for (int i = 0; i < 16; ++i) conk_ws[s0 * 256 + tid + i * 256] = bufC[tid + i * 256];
  __syncthreads();
  gemm_16x256(bufA, packWv, nullptr, bufC, 0);       // v
  __syncthreads();
#pragma unroll
  for (int i = 0; i < 16; ++i) v_ws[s0 * 256 + tid + i * 256] = bufC[tid + i * 256];
  gemm_16x256(bufB, packPk1, pkb1, bufA, 1);         // hidden (relu) -> bufA
  __syncthreads();
  gemm_16x256(bufA, packPk2, pkb2, bufC, 0);         // pos_k
  __syncthreads();
#pragma unroll
  for (int i = 0; i < 16; ++i) posk_ws[s0 * 256 + tid + i * 256] = bufC[tid + i * 256];
}

// ---------------------------------------------------------------------------
// Kernel 3: attention (softmax over G=8 per head) fused with the output
// projection (WMMA) and the residual add.  16 queries / block (36 blocks).
// ---------------------------------------------------------------------------
__global__ __launch_bounds__(256) void attn_out_stage(
    const float* __restrict__ conq_ws, const float* __restrict__ posq_ws,
    const float* __restrict__ conk_ws, const float* __restrict__ posk_ws,
    const float* __restrict__ v_ws,    const float* __restrict__ dec,
    const __bf16* __restrict__ packOut, const float* __restrict__ outb,
    float* __restrict__ out) {
  __shared__ float bufA[16 * 256];
  __shared__ float bufB[16 * 256];
  const int tid = threadIdx.x;
  const int r0  = blockIdx.x * 16;

  if (tid < 128) {                       // one thread per (row, head)
    const int rl = tid >> 3, h = tid & 7;
    const int qr = r0 + rl;
    const float* cq = conq_ws + qr * 256 + h * 32;
    const float* pq = posq_ws + qr * 256 + h * 32;
    float sim[G];
    float mx = -1e30f;
#pragma unroll
    for (int j = 0; j < G; ++j) {
      const float* ck = conk_ws + (qr * G + j) * 256 + h * 32;
      const float* pk = posk_ws + (qr * G + j) * 256 + h * 32;
      float acc = 0.f;
#pragma unroll
      for (int i = 0; i < 32; ++i) acc += cq[i] * ck[i] + pq[i] * pk[i];
      acc *= ATT_SCALE;
      sim[j] = acc;
      mx = fmaxf(mx, acc);
    }
    float den = 0.f;
#pragma unroll
    for (int j = 0; j < G; ++j) { sim[j] = __expf(sim[j] - mx); den += sim[j]; }
    const float inv = 1.0f / den;
#pragma unroll
    for (int i = 0; i < 32; ++i) {
      float acc = 0.f;
#pragma unroll
      for (int j = 0; j < G; ++j) acc += sim[j] * v_ws[(qr * G + j) * 256 + h * 32 + i];
      bufA[rl * 256 + h * 32 + i] = acc * inv;
    }
  }
  __syncthreads();
  gemm_16x256(bufA, packOut, outb, bufB, 0);
  __syncthreads();
#pragma unroll
  for (int i = 0; i < 16; ++i) {
    const int idx = tid + i * 256;
    out[r0 * 256 + idx] = dec[r0 * 256 + idx] + bufB[idx];
  }
}

// ---------------------------------------------------------------------------
// Launcher.  Workspace layout (~16.5 MB, all 256B aligned):
//   bf16 weight packs (8 x 128KB), con_q, pos_q, vgrid, con_k, v, pos_k
// ---------------------------------------------------------------------------
extern "C" void kernel_launch(void* const* d_in, const int* in_sizes, int n_in,
                              void* d_out, int out_size, void* d_ws, size_t ws_size,
                              hipStream_t stream) {
  const float* dec     = (const float*)d_in[0];
  const float* bev     = (const float*)d_in[1];
  const float* qscale  = (const float*)d_in[2];
  const float* refp    = (const float*)d_in[3];
  const float* Wq      = (const float*)d_in[4];
  const float* bq      = (const float*)d_in[5];
  const float* Wk      = (const float*)d_in[6];
  const float* Wv      = (const float*)d_in[7];
  const float* pq_W1   = (const float*)d_in[8];
  const float* pq_b1   = (const float*)d_in[9];
  const float* pq_W2   = (const float*)d_in[10];
  const float* pq_b2   = (const float*)d_in[11];
  const float* pk_W1   = (const float*)d_in[12];
  const float* pk_b1   = (const float*)d_in[13];
  const float* pk_W2   = (const float*)d_in[14];
  const float* pk_b2   = (const float*)d_in[15];
  const float* off_W1  = (const float*)d_in[16];
  const float* off_b1  = (const float*)d_in[17];
  const float* off_W2  = (const float*)d_in[18];
  const float* off_b2  = (const float*)d_in[19];
  const float* out_W   = (const float*)d_in[20];
  const float* out_b   = (const float*)d_in[21];
  float* out = (float*)d_out;

  char* ws = (char*)d_ws;
  size_t off = 0;
  auto give = [&](size_t bytes) -> char* {
    char* p = ws + off;
    off = (off + bytes + 255) & ~(size_t)255;
    return p;
  };
  __bf16* pack   = (__bf16*)give((size_t)8 * 65536 * sizeof(__bf16));
  float*  conq   = (float*)give((size_t)BS * 256 * sizeof(float));
  float*  posq   = (float*)give((size_t)BS * 256 * sizeof(float));
  float*  vgrid  = (float*)give((size_t)NS * 2 * sizeof(float));
  float*  conk   = (float*)give((size_t)NS * 256 * sizeof(float));
  float*  vbuf   = (float*)give((size_t)NS * 256 * sizeof(float));
  float*  posk   = (float*)give((size_t)NS * 256 * sizeof(float));
  (void)ws_size; (void)n_in; (void)in_sizes; (void)out_size;

  // pack slots: 0=Wq 1=pq_W1 2=pq_W2 3=pk_W1 4=pk_W2 5=Wk 6=Wv 7=out_W
  pack_weights_bf16<<<2048, 256, 0, stream>>>(Wq, pq_W1, pq_W2, pk_W1, pk_W2,
                                              Wk, Wv, out_W, pack);
  query_stage<<<BS / 16, 256, 0, stream>>>(
      dec, refp, qscale,
      pack + 0 * 65536, bq,
      off_W1, off_b1, off_W2, off_b2,
      pack + 1 * 65536, pq_b1, pack + 2 * 65536, pq_b2,
      conq, posq, vgrid);
  sample_stage<<<NS / 16, 256, 0, stream>>>(
      bev, vgrid,
      pack + 5 * 65536, pack + 6 * 65536,
      pack + 3 * 65536, pk_b1, pack + 4 * 65536, pk_b2,
      conk, vbuf, posk);
  attn_out_stage<<<BS / 16, 256, 0, stream>>>(
      conq, posq, conk, posk, vbuf, dec,
      pack + 7 * 65536, out_b, out);
}
